// DeepFourierTransform_26259430048287
// MI455X (gfx1250) — compile-verified, matrix operand order checked by reading, not run on hardware
//
#include <hip/hip_runtime.h>
#include <hip/hip_bf16.h>

// ---- shapes from the reference ----
#define SEQ      4096
#define NPERSEG  256
#define HOP      128
#define NWIN     31          // (4096-256)/128 + 1
#define MAPPING  512
#define OUTD     4
#define FEAT     (2*MAPPING) // 1024
#define WELEMS   (MAPPING*NPERSEG)   // 131072 elements per weight matrix

typedef _Float16 h4  __attribute__((ext_vector_type(4)));
typedef _Float16 h8  __attribute__((ext_vector_type(8)));
typedef _Float16 v16h __attribute__((ext_vector_type(16)));
typedef float    f4  __attribute__((ext_vector_type(4)));
typedef float    f8  __attribute__((ext_vector_type(8)));
typedef float    v8f __attribute__((ext_vector_type(8)));

// ---------------------------------------------------------------------------
// Prep: convert w_cos | w_sin (each 512x256 f32) to f16 in d_ws, back to back.
// Runs once per launch; 1 MB read / 512 KB written -> negligible vs 64 MB of x.
// ---------------------------------------------------------------------------
__global__ __launch_bounds__(256)
void cvt_weights_kernel(const float* __restrict__ w_cos,
                        const float* __restrict__ w_sin,
                        _Float16* __restrict__ wh)
{
    int i = blockIdx.x * blockDim.x + threadIdx.x;   // one f4 per thread
    const int n4 = WELEMS / 4;
    if (i < n4) {
        f4 v = ((const f4*)w_cos)[i];
        h4 h; h.x=(_Float16)v.x; h.y=(_Float16)v.y; h.z=(_Float16)v.z; h.w=(_Float16)v.w;
        ((h4*)wh)[i] = h;
    } else {
        i -= n4;
        f4 v = ((const f4*)w_sin)[i];
        h4 h; h.x=(_Float16)v.x; h.y=(_Float16)v.y; h.z=(_Float16)v.z; h.w=(_Float16)v.w;
        ((h4*)(wh + WELEMS))[i] = h;
    }
}

// ---------------------------------------------------------------------------
// Main fused kernel. One block (8 wave32s) per batch row b.
//  - x row -> LDS as f16 once (overlapping windows read in place)
//  - GEMM M=32 (31 windows + pad), N=1024 (cos|sin), K=256 via
//    v_wmma_f32_16x16x32_f16; A-fragments hoisted to VGPRs by the compiler
//  - sin(x) folded to cos(x - pi/2): single branchless transcendental path
//  - fused w_out projection + window-mean + log_softmax
// PRE=true: B fragments loaded directly as f16 from d_ws (no cvt in hot loop)
// ---------------------------------------------------------------------------
template <bool PRE>
__global__ __launch_bounds__(256)
void dft_fused_kernel(const float* __restrict__ x,
                      const float* __restrict__ w_cos, const float* __restrict__ b_cos,
                      const float* __restrict__ w_sin, const float* __restrict__ b_sin,
                      const float* __restrict__ w_out, const float* __restrict__ b_out,
                      const _Float16* __restrict__ wh,   // f16 weights (PRE path)
                      float* __restrict__ out)
{
    __shared__ _Float16 lx[SEQ + 256];   // +256 zero pad for window 31 (pad row)
    __shared__ float    red[8 * OUTD];

    const int tid  = threadIdx.x;
    const int lane = tid & 31;
    const int wid  = tid >> 5;           // 0..7
    const int b    = blockIdx.x;

    // ---- stage x row into LDS as f16 (read x exactly once from HBM) ----
    const float* xrow = x + (size_t)b * SEQ;
    for (int i = tid; i < SEQ / 4; i += 256) {
        f4 v = ((const f4*)xrow)[i];
        h4 h;
        h.x = (_Float16)v.x; h.y = (_Float16)v.y;
        h.z = (_Float16)v.z; h.w = (_Float16)v.w;
        ((h4*)lx)[i] = h;
    }
    lx[SEQ + tid] = (_Float16)0.f;       // zero the pad-window tail
    __syncthreads();

    // waves 0..3 -> cos half (features 0..511), waves 4..7 -> sin half
    const bool      use_sin = (wid >= 4);
    const float*    wmatf   = use_sin ? w_sin : w_cos;          // (512,256) f32
    const _Float16* wmath   = wh + (use_sin ? WELEMS : 0);      // (512,256) f16
    const float*    biasv   = use_sin ? b_sin : b_cos;
    // sin(x) == cos(x - pi/2): shift the bias once, use __cosf everywhere
    const float     phase   = use_sin ? -1.57079632679489662f : 0.0f;

    const int l16     = lane & 15;
    const int hi_half = lane >> 4;             // 0: lanes 0-15, 1: lanes 16-31
    const int k_lo    = hi_half ? 8  : 0;      // 16-bit A/B fragment K offsets
    const int k_hi    = hi_half ? 24 : 16;

    float acc0 = 0.f, acc1 = 0.f, acc2 = 0.f, acc3 = 0.f;

    for (int i = 0; i < 8; ++i) {
        const int nt     = wid * 8 + i;        // N-tile 0..63 over the 1024 features
        const int n_glob = nt * 16 + l16;      // this lane's feature column
        const int n_loc  = n_glob & (MAPPING - 1);

        v8f c0 = {}; v8f c1 = {};
#pragma unroll
        for (int ks = 0; ks < 8; ++ks) {       // K = 256 in steps of 32
            const int kb = ks * 32;

            // A fragments from LDS (hoisted across the n-tile loop by compiler)
            const _Float16* a0p = lx + (size_t)l16        * HOP + kb;  // windows 0..15
            const _Float16* a1p = lx + (size_t)(16 + l16) * HOP + kb;  // windows 16..31
            h8 a0lo = *(const h8*)(a0p + k_lo);
            h8 a0hi = *(const h8*)(a0p + k_hi);
            h8 a1lo = *(const h8*)(a1p + k_lo);
            h8 a1hi = *(const h8*)(a1p + k_hi);
            v16h A0 = __builtin_shufflevector(a0lo, a0hi, 0,1,2,3,4,5,6,7,8,9,10,11,12,13,14,15);
            v16h A1 = __builtin_shufflevector(a1lo, a1hi, 0,1,2,3,4,5,6,7,8,9,10,11,12,13,14,15);

            // B fragment: B[k][n] = w[n][k]; w is (N,K) row-major so 8 contiguous
            // elements per lane.
            v16h B;
            if constexpr (PRE) {
                const _Float16* wrow = wmath + (size_t)n_loc * NPERSEG;
                h8 bl = *(const h8*)(wrow + kb + k_lo);   // one global_load_b128
                h8 bh = *(const h8*)(wrow + kb + k_hi);   // one global_load_b128
                B = __builtin_shufflevector(bl, bh, 0,1,2,3,4,5,6,7,8,9,10,11,12,13,14,15);
            } else {
                const float* wrow = wmatf + (size_t)n_loc * NPERSEG;
                f8 blo = *(const f8*)(wrow + kb + k_lo);
                f8 bhi = *(const f8*)(wrow + kb + k_hi);
                h8 bl = __builtin_convertvector(blo, h8);
                h8 bh = __builtin_convertvector(bhi, h8);
                B = __builtin_shufflevector(bl, bh, 0,1,2,3,4,5,6,7,8,9,10,11,12,13,14,15);
            }

            c0 = __builtin_amdgcn_wmma_f32_16x16x32_f16(false, A0, false, B, (short)0, c0, false, false);
            c1 = __builtin_amdgcn_wmma_f32_16x16x32_f16(false, A1, false, B, (short)0, c1, false, false);
        }

        // ---- fused epilogue: activation + w_out projection + window sum ----
        const float bias = biasv[n_loc] + phase;   // phase folds sin -> cos
        const float wo0 = w_out[0 * FEAT + n_glob];
        const float wo1 = w_out[1 * FEAT + n_glob];
        const float wo2 = w_out[2 * FEAT + n_glob];
        const float wo3 = w_out[3 * FEAT + n_glob];
#pragma unroll
        for (int v = 0; v < 8; ++v) {
            // C/D layout: VGPR v holds M = v (+8 for lanes 16-31); N = lane%16
            float f0 = __cosf(c0[v] + bias);
            float f1 = __cosf(c1[v] + bias);
            // c1 row = 16 + v + 8*hi_half; window 31 is the zero pad -> mask out
            float m1 = (v == 7 && hi_half) ? 0.f : 1.f;
            float s  = f0 + m1 * f1;
            acc0 += s * wo0; acc1 += s * wo1; acc2 += s * wo2; acc3 += s * wo3;
        }
    }

    // ---- wave32 reduction ----
    for (int off = 16; off > 0; off >>= 1) {
        acc0 += __shfl_down(acc0, off, 32);
        acc1 += __shfl_down(acc1, off, 32);
        acc2 += __shfl_down(acc2, off, 32);
        acc3 += __shfl_down(acc3, off, 32);
    }
    if (lane == 0) {
        red[wid * OUTD + 0] = acc0;
        red[wid * OUTD + 1] = acc1;
        red[wid * OUTD + 2] = acc2;
        red[wid * OUTD + 3] = acc3;
    }
    __syncthreads();

    // ---- cross-wave reduce + mean + bias + log_softmax (4 values, scalar) ----
    if (tid == 0) {
        float v[OUTD];
#pragma unroll
        for (int o = 0; o < OUTD; ++o) {
            float s = 0.f;
#pragma unroll
            for (int w = 0; w < 8; ++w) s += red[w * OUTD + o];
            v[o] = s * (1.0f / (float)NWIN) + b_out[o];
        }
        float mx = fmaxf(fmaxf(v[0], v[1]), fmaxf(v[2], v[3]));
        float es = 0.f;
#pragma unroll
        for (int o = 0; o < OUTD; ++o) es += __expf(v[o] - mx);
        float lse = mx + __logf(es);
#pragma unroll
        for (int o = 0; o < OUTD; ++o) out[(size_t)b * OUTD + o] = v[o] - lse;
    }
}

extern "C" void kernel_launch(void* const* d_in, const int* in_sizes, int n_in,
                              void* d_out, int out_size, void* d_ws, size_t ws_size,
                              hipStream_t stream)
{
    const float* x     = (const float*)d_in[0];
    const float* w_cos = (const float*)d_in[1];
    const float* b_cos = (const float*)d_in[2];
    const float* w_sin = (const float*)d_in[3];
    const float* b_sin = (const float*)d_in[4];
    const float* w_out = (const float*)d_in[5];
    const float* b_out = (const float*)d_in[6];
    float* out = (float*)d_out;

    const int B = in_sizes[0] / SEQ;                       // 4096
    const size_t need = (size_t)(2 * WELEMS) * sizeof(_Float16);  // 512 KB

    if (ws_size >= need) {
        _Float16* wh = (_Float16*)d_ws;
        // one f4 per thread over both matrices: 2*131072/4 = 65536 threads
        cvt_weights_kernel<<<(2 * WELEMS / 4 + 255) / 256, 256, 0, stream>>>(
            w_cos, w_sin, wh);
        dft_fused_kernel<true><<<B, 256, 0, stream>>>(
            x, w_cos, b_cos, w_sin, b_sin, w_out, b_out, wh, out);
    } else {
        // fallback: convert f32 weights to f16 on the fly inside the hot loop
        dft_fused_kernel<false><<<B, 256, 0, stream>>>(
            x, w_cos, b_cos, w_sin, b_sin, w_out, b_out, (const _Float16*)d_ws, out);
    }
}
